// Generator_17437567222066
// MI455X (gfx1250) — compile-verified, matrix operand order checked by reading, not run on hardware
//
#include <hip/hip_runtime.h>
#include <hip/hip_bf16.h>
#include <math.h>

#define HH   300
#define H3   900
#define VCAB 19495
#define TENC 512
#define TDEC 2048
#define KP   320      // K (=300) padded to multiple of 32 for WMMA
#define NGP  960      // 900 gates padded to multiple of 64 (block-N tile)
#define BN   64       // block tile N
#define BM   64       // block tile M (4 waves x 16)

typedef __attribute__((ext_vector_type(16))) _Float16 v16h;
typedef __attribute__((ext_vector_type(8)))  _Float16 v8h;
typedef __attribute__((ext_vector_type(8)))  float    v8f;

// ---------------------------------------------------------------- utilities

__global__ void k_zero_f32(float* __restrict__ p, long long n) {
  long long i = (long long)blockIdx.x * blockDim.x + threadIdx.x;
  long long stride = (long long)gridDim.x * blockDim.x;
  for (; i < n; i += stride) p[i] = 0.0f;
}

// gather embeddings -> f16 padded [T][KP]
__global__ void k_embed_f16(const int* __restrict__ ids,
                            const float* __restrict__ emb,
                            _Float16* __restrict__ dst, int T) {
  int idx = blockIdx.x * blockDim.x + threadIdx.x;
  if (idx >= T * KP) return;
  int t = idx / KP, k = idx - t * KP;
  float v = (k < HH) ? emb[(long long)ids[t] * HH + k] : 0.0f;
  dst[idx] = (_Float16)v;
}

// convert weight [N][300] f32 -> padded [Npad][KP] f16 (zero pad)
__global__ void k_cvt_w_f16(const float* __restrict__ src,
                            _Float16* __restrict__ dst, int N, int Npad) {
  int idx = blockIdx.x * blockDim.x + threadIdx.x;
  if (idx >= Npad * KP) return;
  int n = idx / KP, k = idx - n * KP;
  float v = (n < N && k < HH) ? src[(long long)n * HH + k] : 0.0f;
  dst[idx] = (_Float16)v;
}

// Whh [900][300] -> WhhT [300][900] (coalesced recurrent matvec)
__global__ void k_transpose_whh(const float* __restrict__ W, float* __restrict__ WT) {
  int idx = blockIdx.x * blockDim.x + threadIdx.x;
  if (idx >= H3 * HH) return;
  int j = idx / HH, k = idx - j * HH;
  WT[k * H3 + j] = W[idx];
}

// ---------------------------------------------------------------------------
// Blocked WMMA GEMM: C = A @ B.T (+bias), optional per-row tag-predicated store.
// A: [M][KP] f16 (M % 64 == 0). B: [Npad][KP] f16, Npad % 64 == 0, pad rows 0.
// Block = 128 threads (4 waves). Block tile 64x64: B tile (64x320 f16 = 40KB)
// staged in LDS and shared by the 4 waves (stacked along M); each wave keeps
// 4 f32 accumulators (16x64) and reuses its A fragment across 4 WMMAs/k-step.
// ---------------------------------------------------------------------------
__global__ void __launch_bounds__(128)
k_gemm64(const _Float16* __restrict__ A,
         const _Float16* __restrict__ B,
         const float* __restrict__ bias,     // may be null
         const int* __restrict__ tags,       // may be null
         int tagval,
         float* __restrict__ C, int ldc, int Nreal) {
  __shared__ _Float16 Bs[BN * KP];           // 40 KB

  const int bn = blockIdx.x, bm = blockIdx.y;
  const int tid = threadIdx.x;

  // cooperative stage of B tile (fully coalesced 16B copies: 2560 uint4)
  {
    const uint4* src = (const uint4*)(B + (long long)bn * BN * KP);
    uint4* dst = (uint4*)Bs;
#pragma unroll
    for (int i = 0; i < (BN * KP * 2 / 16) / 128; ++i)   // 20 per thread
      dst[tid + i * 128] = src[tid + i * 128];
  }
  __syncthreads();

  const int wave = tid >> 5, lane = tid & 31;
  const int r = lane & 15, half = lane >> 4;

  const _Float16* ap = A + (long long)(bm * BM + wave * 16 + r) * KP + half * 8;
  const _Float16* bp = Bs + r * KP + half * 8;

  v8f acc[4];
#pragma unroll
  for (int s = 0; s < 4; ++s) acc[s] = (v8f){0.f,0.f,0.f,0.f,0.f,0.f,0.f,0.f};

#pragma unroll
  for (int kk = 0; kk < KP; kk += 32) {
    v8h a0 = *(const v8h*)(ap + kk);
    v8h a1 = *(const v8h*)(ap + kk + 16);
    v16h av = __builtin_shufflevector(a0, a1,
                 0,1,2,3,4,5,6,7,8,9,10,11,12,13,14,15);
#pragma unroll
    for (int s = 0; s < 4; ++s) {
      const _Float16* bs = bp + s * 16 * KP;
      v8h b0 = *(const v8h*)(bs + kk);
      v8h b1 = *(const v8h*)(bs + kk + 16);
      v16h bv = __builtin_shufflevector(b0, b1,
                   0,1,2,3,4,5,6,7,8,9,10,11,12,13,14,15);
      acc[s] = __builtin_amdgcn_wmma_f32_16x16x32_f16(false, av, false, bv,
                                                      (short)0, acc[s], false, false);
    }
  }

  const int m_base = bm * BM + wave * 16;
#pragma unroll
  for (int s = 0; s < 4; ++s) {
    int n = bn * BN + s * 16 + r;
    if (n < Nreal) {
      float bb = bias ? bias[n] : 0.0f;
#pragma unroll
      for (int v = 0; v < 8; ++v) {
        int m = m_base + v + half * 8;
        bool st = tags ? (tags[m] == tagval) : true;
        if (st) C[(long long)m * ldc + n] = acc[s][v] + bb;
      }
    }
  }
}

// ------------------------------------------------------ sequential GRU cells

__device__ __forceinline__ float sig_(float x) { return 1.0f / (1.0f + expf(-x)); }

// encoder: one block per sequence, h0 = 0, writes final h
__global__ void __launch_bounds__(1024)
k_gru_enc(const float* __restrict__ xg,     // [4*TENC][900], row = seq*TENC+t
          const float* __restrict__ WhhT,   // [300][900]
          const float* __restrict__ bhh,    // [900]
          float* __restrict__ hout) {       // [4][300]
  __shared__ float h[HH + 4];
  __shared__ float gh[H3];
  int tid = threadIdx.x;
  int seq = blockIdx.x;
  if (tid < HH) h[tid] = 0.0f;
  __syncthreads();
  for (int t = 0; t < TENC; ++t) {
    if (tid < H3) {
      float a0 = 0.f, a1 = 0.f, a2 = 0.f, a3 = 0.f;
      for (int k = 0; k < HH; k += 4) {
        a0 += WhhT[(k + 0) * H3 + tid] * h[k + 0];
        a1 += WhhT[(k + 1) * H3 + tid] * h[k + 1];
        a2 += WhhT[(k + 2) * H3 + tid] * h[k + 2];
        a3 += WhhT[(k + 3) * H3 + tid] * h[k + 3];
      }
      gh[tid] = bhh[tid] + ((a0 + a1) + (a2 + a3));
    }
    __syncthreads();
    if (tid < HH) {
      const float* xr = xg + (long long)(seq * TENC + t) * H3;
      float rr = sig_(xr[tid] + gh[tid]);
      float zz = sig_(xr[HH + tid] + gh[HH + tid]);
      float nn = tanhf(xr[2 * HH + tid] + rr * gh[2 * HH + tid]);
      h[tid] = (1.0f - zz) * nn + zz * h[tid];
    }
    __syncthreads();
  }
  if (tid < HH) hout[seq * HH + tid] = h[tid];
}

// hidden = concat(h4)[1200] @ state_W.T + state_b
__global__ void k_state(const float* __restrict__ h4,   // [1200]
                        const float* __restrict__ W,    // [300][1200]
                        const float* __restrict__ b,
                        float* __restrict__ hidden) {
  int j = threadIdx.x;
  if (j >= HH) return;
  float acc = b[j];
  const float* wr = W + (long long)j * (4 * HH);
  for (int k = 0; k < 4 * HH; ++k) acc += wr[k] * h4[k];
  hidden[j] = acc;
}

// decoder: single block, 2048 steps, writes padded f16 hidden states
__global__ void __launch_bounds__(1024)
k_gru_dec(const float* __restrict__ xg,     // [TDEC][900]
          const float* __restrict__ WhhT,   // [300][900]
          const float* __restrict__ bhh,    // [900]
          const float* __restrict__ h0,     // [300]
          _Float16* __restrict__ hs) {      // [TDEC][KP] f16 (K-padded)
  __shared__ float h[HH + 4];
  __shared__ float gh[H3];
  int tid = threadIdx.x;
  if (tid < HH) h[tid] = h0[tid];
  __syncthreads();
  for (int t = 0; t < TDEC; ++t) {
    if (tid < H3) {
      float a0 = 0.f, a1 = 0.f, a2 = 0.f, a3 = 0.f;
      for (int k = 0; k < HH; k += 4) {
        a0 += WhhT[(k + 0) * H3 + tid] * h[k + 0];
        a1 += WhhT[(k + 1) * H3 + tid] * h[k + 1];
        a2 += WhhT[(k + 2) * H3 + tid] * h[k + 2];
        a3 += WhhT[(k + 3) * H3 + tid] * h[k + 3];
      }
      gh[tid] = bhh[tid] + ((a0 + a1) + (a2 + a3));
    }
    __syncthreads();
    if (tid < HH) {
      const float* xr = xg + (long long)t * H3;
      float rr = sig_(xr[tid] + gh[tid]);
      float zz = sig_(xr[HH + tid] + gh[HH + tid]);
      float nn = tanhf(xr[2 * HH + tid] + rr * gh[2 * HH + tid]);
      h[tid] = (1.0f - zz) * nn + zz * h[tid];
    }
    __syncthreads();
    if (tid < KP) hs[(long long)t * KP + tid] = (tid < HH) ? (_Float16)h[tid] : (_Float16)0.0f;
  }
}

// ---------------------------------------------------------------- launcher

extern "C" void kernel_launch(void* const* d_in, const int* in_sizes, int n_in,
                              void* d_out, int out_size, void* d_ws, size_t ws_size,
                              hipStream_t stream) {
  const int*   verb_in   = (const int*)d_in[0];
  const int*   pron_in   = (const int*)d_in[1];
  const int*   noun_in   = (const int*)d_in[2];
  const int*   sen_in    = (const int*)d_in[3];
  const int*   tgt_in    = (const int*)d_in[4];
  const int*   tgt_tag   = (const int*)d_in[5];
  const float* emb       = (const float*)d_in[6];
  const float* enc_Wih   = (const float*)d_in[7];
  const float* enc_Whh   = (const float*)d_in[8];
  const float* enc_bih   = (const float*)d_in[9];
  const float* enc_bhh   = (const float*)d_in[10];
  const float* dec_Wih   = (const float*)d_in[11];
  const float* dec_Whh   = (const float*)d_in[12];
  const float* dec_bih   = (const float*)d_in[13];
  const float* dec_bhh   = (const float*)d_in[14];
  const float* state_W   = (const float*)d_in[15];
  const float* state_b   = (const float*)d_in[16];
  const float* noun_W    = (const float*)d_in[17];
  const float* noun_b    = (const float*)d_in[18];
  const float* verb_W    = (const float*)d_in[19];
  const float* verb_b    = (const float*)d_in[20];
  const float* pron_W    = (const float*)d_in[21];
  const float* pron_b    = (const float*)d_in[22];
  const float* sen_W     = (const float*)d_in[23];
  const float* sen_b     = (const float*)d_in[24];
  float* out = (float*)d_out;

  // ---- workspace carve (256B aligned)
  char* ws = (char*)d_ws;
  size_t off = 0;
  auto alloc = [&](size_t bytes) -> void* {
    off = (off + 255) & ~(size_t)255;
    void* p = ws + off;
    off += bytes;
    return p;
  };
  _Float16* enc_x   = (_Float16*)alloc((size_t)4 * TENC * KP * 2);
  _Float16* dec_x   = (_Float16*)alloc((size_t)TDEC * KP * 2);
  _Float16* wih_e   = (_Float16*)alloc((size_t)NGP * KP * 2);
  _Float16* wih_d   = (_Float16*)alloc((size_t)NGP * KP * 2);
  float*    whhT_e  = (float*)   alloc((size_t)HH * H3 * 4);
  float*    whhT_d  = (float*)   alloc((size_t)HH * H3 * 4);
  float*    xg_e    = (float*)   alloc((size_t)4 * TENC * H3 * 4);
  float*    xg_d    = (float*)   alloc((size_t)TDEC * H3 * 4);
  float*    h4      = (float*)   alloc((size_t)4 * HH * 4);
  float*    hidden0 = (float*)   alloc((size_t)HH * 4);
  _Float16* hs16    = (_Float16*)alloc((size_t)TDEC * KP * 2);
  const int NOUN_P = 10048, VERB_P = 6592, PRON_P = 128, SEN_P = 19520; // %64==0
  _Float16* nw16 = (_Float16*)alloc((size_t)NOUN_P * KP * 2);
  _Float16* vw16 = (_Float16*)alloc((size_t)VERB_P * KP * 2);
  _Float16* pw16 = (_Float16*)alloc((size_t)PRON_P * KP * 2);
  _Float16* sw16 = (_Float16*)alloc((size_t)SEN_P * KP * 2);

  auto cdiv = [](long long a, long long b) -> int { return (int)((a + b - 1) / b); };

  // 1) zero output (heads pad with zeros past their width)
  k_zero_f32<<<4096, 256, 0, stream>>>(out, (long long)TDEC * VCAB);

  // 2) embeddings -> padded f16
  k_embed_f16<<<cdiv((long long)TENC * KP, 256), 256, 0, stream>>>(verb_in, emb, enc_x + 0 * TENC * KP, TENC);
  k_embed_f16<<<cdiv((long long)TENC * KP, 256), 256, 0, stream>>>(pron_in, emb, enc_x + 1 * TENC * KP, TENC);
  k_embed_f16<<<cdiv((long long)TENC * KP, 256), 256, 0, stream>>>(noun_in, emb, enc_x + 2 * TENC * KP, TENC);
  k_embed_f16<<<cdiv((long long)TENC * KP, 256), 256, 0, stream>>>(sen_in,  emb, enc_x + 3 * TENC * KP, TENC);
  k_embed_f16<<<cdiv((long long)TDEC * KP, 256), 256, 0, stream>>>(tgt_in,  emb, dec_x, TDEC);

  // 3) convert/pad weights to f16
  k_cvt_w_f16<<<cdiv((long long)NGP * KP, 256), 256, 0, stream>>>(enc_Wih, wih_e, H3, NGP);
  k_cvt_w_f16<<<cdiv((long long)NGP * KP, 256), 256, 0, stream>>>(dec_Wih, wih_d, H3, NGP);
  k_cvt_w_f16<<<cdiv((long long)NOUN_P * KP, 256), 256, 0, stream>>>(noun_W, nw16, 10000, NOUN_P);
  k_cvt_w_f16<<<cdiv((long long)VERB_P * KP, 256), 256, 0, stream>>>(verb_W, vw16, 6555, VERB_P);
  k_cvt_w_f16<<<cdiv((long long)PRON_P * KP, 256), 256, 0, stream>>>(pron_W, pw16, 82, PRON_P);
  k_cvt_w_f16<<<cdiv((long long)SEN_P * KP, 256), 256, 0, stream>>>(sen_W, sw16, VCAB, SEN_P);

  // 4) transpose Whh for coalesced recurrent matvec
  k_transpose_whh<<<cdiv((long long)H3 * HH, 256), 256, 0, stream>>>(enc_Whh, whhT_e);
  k_transpose_whh<<<cdiv((long long)H3 * HH, 256), 256, 0, stream>>>(dec_Whh, whhT_d);

  // 5) WMMA input-gate GEMMs: xg = x @ Wih.T + bih   (M=2048, N=900, K=320)
  {
    dim3 grid(NGP / BN, (4 * TENC) / BM);
    k_gemm64<<<grid, 128, 0, stream>>>(enc_x, wih_e, enc_bih, nullptr, -1,
                                       xg_e, H3, H3);
    dim3 grid2(NGP / BN, TDEC / BM);
    k_gemm64<<<grid2, 128, 0, stream>>>(dec_x, wih_d, dec_bih, nullptr, -1,
                                        xg_d, H3, H3);
  }

  // 6) 4 encoder GRUs in parallel (one block each)
  k_gru_enc<<<4, 1024, 0, stream>>>(xg_e, whhT_e, enc_bhh, h4);

  // 7) decoder initial state
  k_state<<<1, 320, 0, stream>>>(h4, state_W, state_b, hidden0);

  // 8) decoder GRU (sequential, single block) -> padded f16 hidden states
  k_gru_dec<<<1, 1024, 0, stream>>>(xg_d, whhT_d, dec_bhh, hidden0, hs16);

  // 9) WMMA head GEMMs, tag-predicated stores + bias
  struct Head { const _Float16* W; const float* b; int Npad, Nreal, tag; };
  Head heads[4] = {
    { nw16, noun_b, NOUN_P, 10000, 0 },
    { vw16, verb_b, VERB_P, 6555,  1 },
    { pw16, pron_b, PRON_P, 82,    2 },
    { sw16, sen_b,  SEN_P,  VCAB,  3 },
  };
  for (int i = 0; i < 4; ++i) {
    dim3 grid(heads[i].Npad / BN, TDEC / BM);
    k_gemm64<<<grid, 128, 0, stream>>>(hs16, heads[i].W, heads[i].b,
                                       tgt_tag, heads[i].tag, out,
                                       VCAB, heads[i].Nreal);
  }
}
